// SimVQ_41077067219309
// MI455X (gfx1250) — matrix operand order repaired and checked by reading, not compiled, not run on hardware
//
#include <hip/hip_runtime.h>
#include <hip/hip_bf16.h>

#define NUM_EMB 16384
#define EMB_DIM 64
#define BT      16384          // 16 * 1024 rows of z
#define CHUNK_N 128            // codebook rows staged in LDS per iteration
#define CHUNK_BYTES (CHUNK_N * EMB_DIM * 2)

typedef __attribute__((ext_vector_type(16))) _Float16 v16h;
typedef __attribute__((ext_vector_type(8)))  _Float16 v8h;
typedef __attribute__((ext_vector_type(8)))  float    v8f;

#if defined(__has_builtin)
#if __has_builtin(__builtin_amdgcn_tensor_load_to_lds) && \
    __has_builtin(__builtin_amdgcn_s_wait_tensorcnt)
#define USE_TDM 1
#endif
#endif

#ifdef USE_TDM
typedef unsigned int u32x4 __attribute__((ext_vector_type(4)));
typedef int          i32x4 __attribute__((ext_vector_type(4)));
typedef int          i32x8 __attribute__((ext_vector_type(8)));

// Issue a TDM load of a 2D tile: CHUNK_N rows x 64 f16 from gsrc into LDS.
// Descriptor per CDNA5 ISA ch.8 (D# group0/group1; groups 2-3 zero => 2D).
__device__ __forceinline__ void tdm_stage(const _Float16* gsrc, unsigned lds_off) {
    unsigned long long ga = (unsigned long long)gsrc;
    u32x4 g0;
    g0[0] = 1u;                                          // count=1 (valid user D#)
    g0[1] = lds_off;                                     // lds_addr (bytes)
    g0[2] = (unsigned)(ga & 0xffffffffu);                // global_addr[31:0]
    g0[3] = (unsigned)((ga >> 32) & 0x1ffffffu)          // global_addr[56:32]
          | (2u << 30);                                  // type=2 ("image")
    i32x8 g1;
    g1[0] = (1 << 16);                                   // data_size=1 (2 bytes/elt)
    g1[1] = (EMB_DIM << 16);                             // tensor_dim0=64  @bits[79:48]
    g1[2] = (int)(((unsigned)NUM_EMB & 0xffffu) << 16);  // tensor_dim1     @bits[111:80]
    g1[3] = (EMB_DIM << 16);                             // tile_dim0=64    @bits[127:112]
    g1[4] = CHUNK_N;                                     // tile_dim1=128   @bits[143:128]
    g1[5] = EMB_DIM;                                     // dim0_stride=64  @bits[207:160]
    g1[6] = 0;
    g1[7] = 0;
    i32x4 gz = {0, 0, 0, 0};
#if defined(__clang_major__) && __clang_major__ >= 23
    i32x8 gz8 = {0, 0, 0, 0, 0, 0, 0, 0};
    __builtin_amdgcn_tensor_load_to_lds(g0, g1, gz, gz, gz8, 0);
#else
    __builtin_amdgcn_tensor_load_to_lds(g0, g1, gz, gz, 0);
#endif
}
#endif

// ---------------------------------------------------------------------------
// Kernel 0: zero the loss accumulator slot (graph-replay safe re-init)
// ---------------------------------------------------------------------------
__global__ void vq_init_loss(float* loss) { *loss = 0.0f; }

// ---------------------------------------------------------------------------
// Kernel 1: quant_codebook = codebook @ W^T + b, then L2-normalize rows.
// ---------------------------------------------------------------------------
__global__ __launch_bounds__(64) void vq_proj_norm_cb(
    const float* __restrict__ codebook, const float* __restrict__ proj_w,
    const float* __restrict__ proj_b,
    float* __restrict__ qc, _Float16* __restrict__ cn)
{
    __shared__ float srow[EMB_DIM];
    __shared__ float ssq[EMB_DIM];
    const int n = blockIdx.x;
    const int d = threadIdx.x;

    srow[d] = codebook[n * EMB_DIM + d];
    __syncthreads();

    float acc = proj_b[d];
    const float* wrow = proj_w + d * EMB_DIM;   // qc[n,d] = sum_k cb[n,k] * W[d,k]
#pragma unroll 8
    for (int k = 0; k < EMB_DIM; ++k) acc = fmaf(srow[k], wrow[k], acc);

    qc[n * EMB_DIM + d] = acc;
    ssq[d] = acc * acc;
    __syncthreads();
#pragma unroll
    for (int s = 32; s > 0; s >>= 1) {
        if (d < s) ssq[d] += ssq[d + s];
        __syncthreads();
    }
    const float inv = 1.0f / fmaxf(sqrtf(ssq[0]), 1e-12f);
    cn[n * EMB_DIM + d] = (_Float16)(acc * inv);
}

// ---------------------------------------------------------------------------
// Kernel 2: L2-normalize z rows, emit f16 (WMMA A operand feed).
// ---------------------------------------------------------------------------
__global__ __launch_bounds__(64) void vq_norm_z(
    const float* __restrict__ z, _Float16* __restrict__ zn)
{
    __shared__ float ssq[EMB_DIM];
    const int n = blockIdx.x;
    const int d = threadIdx.x;
    const float v = z[n * EMB_DIM + d];
    ssq[d] = v * v;
    __syncthreads();
#pragma unroll
    for (int s = 32; s > 0; s >>= 1) {
        if (d < s) ssq[d] += ssq[d + s];
        __syncthreads();
    }
    const float inv = 1.0f / fmaxf(sqrtf(ssq[0]), 1e-12f);
    zn[n * EMB_DIM + d] = (_Float16)(v * inv);
}

// ---------------------------------------------------------------------------
// Kernel 3: fused similarity-GEMM + row-argmax.
// Each wave owns an M=16 tile of zn (A in registers for the whole N sweep) and
// scans all 16384 codebook rows with v_wmma_f32_16x16x32_f16. B tiles are
// double-buffered in LDS and staged by the Tensor Data Mover (TENSORcnt),
// overlapping DMA with WMMA issue; only wave 0 drives the TDM, a workgroup
// barrier publishes the tile to all 8 waves.
// ---------------------------------------------------------------------------
__global__ __launch_bounds__(256) void vq_argmax_wmma(
    const _Float16* __restrict__ zn,   // [BT][64]
    const _Float16* __restrict__ cn,   // [NUM_EMB][64]
    int* __restrict__ out_idx)         // [BT]
{
    __shared__ _Float16 sB[2][CHUNK_N * EMB_DIM];   // 2 x 16 KB
    const int tid    = threadIdx.x;
    const int lane   = tid & 31;
    const int wave   = tid >> 5;
    const int m_tile = blockIdx.x * 8 + wave;

    // ---- A operand: zn rows m_tile*16 .. +15, K=0..63 (two v16h) ----------
    // lanes 0-15:  M=lane,    K = {0..15} (+32 for second half)
    // lanes 16-31: M=lane-16, K = {16..31} (+32)
    const int mrow = lane & 15;
    const int khi  = (lane >= 16) ? 16 : 0;
    const _Float16* zrow = zn + (m_tile * 16 + mrow) * EMB_DIM;
    v16h a0, a1;
    {
        v8h lo0 = *(const v8h*)(zrow + khi);
        v8h hi0 = *(const v8h*)(zrow + khi + 8);
        v8h lo1 = *(const v8h*)(zrow + 32 + khi);
        v8h hi1 = *(const v8h*)(zrow + 32 + khi + 8);
#pragma unroll
        for (int e = 0; e < 8; ++e) {
            a0[e] = lo0[e]; a0[e + 8] = hi0[e];
            a1[e] = lo1[e]; a1[e + 8] = hi1[e];
        }
    }

    float best[8];
    int   bidx[8];
#pragma unroll
    for (int r = 0; r < 8; ++r) { best[r] = -3.4e38f; bidx[r] = 0; }

    const int ncol = lane & 15;                 // B column this lane owns
    const int kb   = (lane >= 16) ? 16 : 0;     // B K-group base

#ifdef USE_TDM
    const unsigned lds_b0 = (unsigned)(unsigned long long)&sB[0][0];
    const unsigned lds_b1 = (unsigned)(unsigned long long)&sB[1][0];
    if (wave == 0) tdm_stage(cn, lds_b0);                 // prologue: chunk 0
#else
    {   // cooperative prologue stage of chunk 0 into buffer 0
        const int4* g = (const int4*)cn;
        int4* s = (int4*)&sB[0][0];
#pragma unroll
        for (int j = 0; j < 4; ++j) s[tid * 4 + j] = g[tid * 4 + j];
    }
#endif

    for (int chunk = 0, it = 0; chunk < NUM_EMB; chunk += CHUNK_N, ++it) {
        const int b = it & 1;
#ifdef USE_TDM
        if (wave == 0) __builtin_amdgcn_s_wait_tensorcnt(0);  // buf b ready
#endif
        __syncthreads();    // publish buf b; all waves done reading buf b^1
        if (chunk + CHUNK_N < NUM_EMB) {
#ifdef USE_TDM
            if (wave == 0)
                tdm_stage(cn + (chunk + CHUNK_N) * EMB_DIM, b ? lds_b0 : lds_b1);
#else
            const int4* g = (const int4*)(cn + (chunk + CHUNK_N) * EMB_DIM);
            int4* s = (int4*)&sB[b ^ 1][0];
#pragma unroll
            for (int j = 0; j < 4; ++j) s[tid * 4 + j] = g[tid * 4 + j];
#endif
        }

#pragma unroll
        for (int t = 0; t < CHUNK_N / 16; ++t) {
            const _Float16* brow = &sB[b][(t * 16 + ncol) * EMB_DIM + kb];
            v16h b0 = *(const v16h*)(brow);        // K = 0..31 slice
            v16h b1 = *(const v16h*)(brow + 32);   // K = 32..63 slice
            v8f c = {};
            c = __builtin_amdgcn_wmma_f32_16x16x32_f16(
                    false, a0, false, b0, (short)0, c, false, false);
            c = __builtin_amdgcn_wmma_f32_16x16x32_f16(
                    false, a1, false, b1, (short)0, c, false, false);
            const int n = chunk + t * 16 + ncol;   // candidate index, constant/lane
#pragma unroll
            for (int r = 0; r < 8; ++r) {
                if (c[r] > best[r]) { best[r] = c[r]; bidx[r] = n; }  // strict > => first max
            }
        }
    }

    // Butterfly reduce across the 16-lane halves (rows 0-7 in lanes 0-15,
    // rows 8-15 in lanes 16-31). Ties -> lower index (argmin-first semantics).
#pragma unroll
    for (int r = 0; r < 8; ++r) {
#pragma unroll
        for (int s = 1; s < 16; s <<= 1) {
            float ov = __shfl_xor(best[r], s, 32);
            int   oi = __shfl_xor(bidx[r], s, 32);
            if (ov > best[r] || (ov == best[r] && oi < bidx[r])) {
                best[r] = ov; bidx[r] = oi;
            }
        }
    }
    if (lane == 0 || lane == 16) {
        const int mbase = m_tile * 16 + ((lane >> 4) ? 8 : 0);
#pragma unroll
        for (int r = 0; r < 8; ++r) out_idx[mbase + r] = bidx[r];
    }
}

// ---------------------------------------------------------------------------
// Kernel 4: gather quantized rows, emit straight-through output, indices, and
// vq_loss = 1.25 * mean((q - z)^2)  (commitment + beta*codebook collapse).
// ---------------------------------------------------------------------------
__global__ __launch_bounds__(256) void vq_gather_out(
    const float* __restrict__ z, const float* __restrict__ qc,
    const int* __restrict__ idx,
    float* __restrict__ out_q, float* __restrict__ out_loss,
    int* __restrict__ out_idx)
{
    __shared__ float red[256];
    const int gid = blockIdx.x * 256 + threadIdx.x;
    const int row = gid >> 6;
    const int d   = gid & 63;
    const int id  = idx[row];
    const float q = qc[id * EMB_DIM + d];
    out_q[gid] = q;                      // quantized_st == quantized numerically
    if (d == 0) out_idx[row] = id;
    const float diff = q - z[gid];
    red[threadIdx.x] = diff * diff;
    __syncthreads();
#pragma unroll
    for (int s = 128; s > 0; s >>= 1) {
        if (threadIdx.x < s) red[threadIdx.x] += red[threadIdx.x + s];
        __syncthreads();
    }
    if (threadIdx.x == 0)
        atomicAdd(out_loss, red[0] * (1.25f / (float)(BT * EMB_DIM)));
}

// ---------------------------------------------------------------------------
extern "C" void kernel_launch(void* const* d_in, const int* in_sizes, int n_in,
                              void* d_out, int out_size, void* d_ws, size_t ws_size,
                              hipStream_t stream) {
    const float* z        = (const float*)d_in[0];  // [16,1024,64]
    const float* codebook = (const float*)d_in[1];  // [16384,64]
    const float* proj_w   = (const float*)d_in[2];  // [64,64]
    const float* proj_b   = (const float*)d_in[3];  // [64]
    // d_in[4] = scale: positive constant, invariant under argmin -> unused.

    char* ws = (char*)d_ws;
    float*    qc   = (float*)ws;                        // 4 MB  f32 projected codebook
    _Float16* cn   = (_Float16*)(ws + (4 << 20));       // 2 MB  f16 normalized codebook
    _Float16* zn   = (_Float16*)(ws + (6 << 20));       // 2 MB  f16 normalized z
    int*      widx = (int*)(ws + (8 << 20));            // 64 KB argmax indices

    float* out_q    = (float*)d_out;                    // [1048576] quantized_st
    float* out_loss = out_q + (BT * EMB_DIM);           // [1] vq_loss
    int*   out_idx  = (int*)(out_q + BT * EMB_DIM + 1); // [16384] indices

    vq_init_loss   <<<1, 1, 0, stream>>>(out_loss);
    vq_proj_norm_cb<<<NUM_EMB, 64, 0, stream>>>(codebook, proj_w, proj_b, qc, cn);
    vq_norm_z      <<<BT, 64, 0, stream>>>(z, zn);
    vq_argmax_wmma <<<BT / 16 / 8, 256, 0, stream>>>(zn, cn, widx);
    vq_gather_out  <<<(BT * EMB_DIM) / 256, 256, 0, stream>>>(
        z, qc, widx, out_q, out_loss, out_idx);
}
